// OptimizedCNN_QNN_Model_22033182228578
// MI455X (gfx1250) — matrix operand order repaired and checked by reading, not compiled, not run on hardware
//
#include <hip/hip_runtime.h>

typedef __attribute__((ext_vector_type(16))) _Float16 v16h;
typedef __attribute__((ext_vector_type(8)))  float    v8f;

#define BN_EPS 1e-5f

// ---------------------------------------------------------------------------
// conv1: 1->12, k5 s2 p2, 64x64 -> 32x32, fused BN+ReLU.
// Writes f16 in [b][oh*32+ow][ci] layout (gather-friendly for conv2 WMMA).
// ---------------------------------------------------------------------------
__global__ void conv1_bn_relu_f16(const float* __restrict__ x, const float* __restrict__ w,
                                  const float* __restrict__ cb, const float* __restrict__ g,
                                  const float* __restrict__ bb, const float* __restrict__ m,
                                  const float* __restrict__ vv, _Float16* __restrict__ out) {
    int i = blockIdx.x * 256 + threadIdx.x;               // 1024*12*32*32
    if (i >= 1024 * 12 * 32 * 32) return;
    int ow = i & 31; int t = i >> 5; int oh = t & 31; t >>= 5;
    int co = t % 12; int b = t / 12;
    const float* xb = x + (size_t)b * 4096;
    const float* wc = w + co * 25;
    float acc = 0.f;
#pragma unroll
    for (int kh = 0; kh < 5; ++kh) {
        int ih = oh * 2 - 2 + kh;
        if ((unsigned)ih < 64u) {
#pragma unroll
            for (int kw = 0; kw < 5; ++kw) {
                int iw = ow * 2 - 2 + kw;
                if ((unsigned)iw < 64u) acc += xb[ih * 64 + iw] * wc[kh * 5 + kw];
            }
        }
    }
    float iv = g[co] * rsqrtf(vv[co] + BN_EPS);
    float v = fmaxf((acc + cb[co]) * iv + (bb[co] - m[co] * iv), 0.f);
    out[((size_t)b * 1024 + oh * 32 + ow) * 12 + co] = (_Float16)v;
}

// ---------------------------------------------------------------------------
// implicit-im2col gather for 3x3 pad-1 convs, activations f16 [spatial][CIN]
// ---------------------------------------------------------------------------
template <int CIN, int HIN, int STRIDE, int K>
__device__ inline _Float16 convA(const _Float16* __restrict__ xb, int oh, int ow, int k) {
    int ci = k / 9;
    int r  = k - ci * 9;
    int kh = r / 3;
    int kw = r - kh * 3;
    int ih = oh * STRIDE - 1 + kh;
    int iw = ow * STRIDE - 1 + kw;
    bool ok = ((unsigned)ih < (unsigned)HIN) && ((unsigned)iw < (unsigned)HIN) && (k < K);
    int idx = ok ? (ih * HIN + iw) * CIN + ci : 0;
    _Float16 v = xb[idx];
    return ok ? v : (_Float16)0.f;
}

// ---------------------------------------------------------------------------
// Implicit-GEMM WMMA conv (3x3, pad 1) + BN + ReLU.
// M = 1024*HOUT*HOUT spatial rows, N = COUTP (zero-padded weights), K padded.
// One wave computes a 16x16 output tile via v_wmma_f32_16x16x32_f16.
// Output: [m][COUT] (f16 or f32).
// ---------------------------------------------------------------------------
template <int CIN, int HIN, int STRIDE, int HOUT, int K, int KP, int COUT, int COUTP, bool F16OUT>
__global__ void conv_wmma_bn_relu(const _Float16* __restrict__ A, const _Float16* __restrict__ W,
                                  const float* __restrict__ cb, const float* __restrict__ g,
                                  const float* __restrict__ bb, const float* __restrict__ m,
                                  const float* __restrict__ vv, void* __restrict__ out) {
    constexpr int TILES_N = COUTP / 16;
    constexpr int M = 1024 * HOUT * HOUT;
    int wave = threadIdx.x >> 5;
    int lane = threadIdx.x & 31;
    int tile = blockIdx.x * 8 + wave;
    if (tile >= (M / 16) * TILES_N) return;
    int tM = tile / TILES_N;
    int tN = tile % TILES_N;
    int row = lane & 15;
    int hi  = lane >> 4;

    int mrow = tM * 16 + row;
    int b   = mrow / (HOUT * HOUT);
    int pos = mrow % (HOUT * HOUT);
    int oh = pos / HOUT, ow = pos % HOUT;
    const _Float16* xb   = A + (size_t)b * HIN * HIN * CIN;
    const _Float16* brow = W + (tN * 16 + row) * KP;

    v8f acc = {};
#pragma unroll
    for (int kk = 0; kk < KP; kk += 32) {
        v16h av, bv;
#pragma unroll
        for (int t = 0; t < 8; ++t) {
            av[t]     = convA<CIN, HIN, STRIDE, K>(xb, oh, ow, kk + hi * 8 + t);
            av[t + 8] = convA<CIN, HIN, STRIDE, K>(xb, oh, ow, kk + 16 + hi * 8 + t);
        }
#pragma unroll
        for (int t = 0; t < 16; ++t) bv[t] = brow[kk + hi * 8 + (t & 7) + ((t >> 3) << 4)];
        acc = __builtin_amdgcn_wmma_f32_16x16x32_f16(false, av, false, bv,
                                                     (short)0, acc, false, false);
    }
    // D layout: lane-half hi, vgpr t -> M = hi*8+t ; N = lane&15
    int co = tN * 16 + row;
    if (co < COUT) {
        float iv  = g[co] * rsqrtf(vv[co] + BN_EPS);
        float sh  = bb[co] - m[co] * iv;
        float cbc = cb[co];
#pragma unroll
        for (int t = 0; t < 8; ++t) {
            int mm = tM * 16 + hi * 8 + t;
            float v = fmaxf((acc[t] + cbc) * iv + sh, 0.f);
            if (F16OUT) ((_Float16*)out)[(size_t)mm * COUT + co] = (_Float16)v;
            else        ((float*)out)[(size_t)mm * COUT + co] = v;
        }
    }
}

// ---------------------------------------------------------------------------
// pack conv weights [COUT][K] f32 -> [COUTP][KP] f16, zero padded
// ---------------------------------------------------------------------------
__global__ void pack_convw_f16(const float* __restrict__ src, _Float16* __restrict__ dst,
                               int K, int KP, int COUT, int total) {
    int i = blockIdx.x * 256 + threadIdx.x;
    if (i >= total) return;
    int co = i / KP; int k = i % KP;
    dst[i] = (co < COUT && k < K) ? (_Float16)src[co * K + k] : (_Float16)0.f;
}

// ---------------------------------------------------------------------------
// adaptive avg pool 16x16 -> 3x3 (bins [0,6)[5,11)[10,16)) over conv3 output
// stored as [b*256 + r*16 + c][48]; flatten, f16, pad K 432 -> 448.
// ---------------------------------------------------------------------------
__global__ void pool_flatten_f16(const float* __restrict__ h, _Float16* __restrict__ featH) {
    int i = blockIdx.x * 256 + threadIdx.x;               // 1024*448
    if (i >= 1024 * 448) return;
    int j = i % 448; int b = i / 448;
    if (j >= 432) { featH[i] = (_Float16)0.f; return; }
    int c = j / 9; int jj = j % 9; int r3 = jj / 3; int c3 = jj % 3;
    const int st[3] = {0, 5, 10};
    float s = 0.f;
#pragma unroll
    for (int r = 0; r < 6; ++r)
#pragma unroll
        for (int cc = 0; cc < 6; ++cc)
            s += h[((size_t)b * 256 + (st[r3] + r) * 16 + (st[c3] + cc)) * 48 + c];
    featH[i] = (_Float16)(s * (1.f / 36.f));
}

// ---------------------------------------------------------------------------
// f32 -> f16 row conversion with K padding (fc weights, WMMA B operand)
// ---------------------------------------------------------------------------
__global__ void cvt_f16_pad(const float* __restrict__ src, _Float16* __restrict__ dst,
                            int K, int Kp, int total) {
    int i = blockIdx.x * 256 + threadIdx.x;
    if (i >= total) return;
    int r = i / Kp; int k = i % Kp;
    dst[i] = (k < K) ? (_Float16)src[r * K + k] : (_Float16)0.f;
}

// ---------------------------------------------------------------------------
// WMMA GEMM: C[M,N] = relu(A[M,Kp](f16) * W[N,Kp]^T(f16) + bias)
// ---------------------------------------------------------------------------
template <bool STORE_F16>
__global__ void gemm_wmma_relu(const _Float16* __restrict__ A, const _Float16* __restrict__ W,
                               const float* __restrict__ bias, void* __restrict__ out,
                               int M, int N, int Kp) {
    int wave = threadIdx.x >> 5;
    int lane = threadIdx.x & 31;
    int tile = blockIdx.x * 8 + wave;
    int tilesN = N >> 4;
    int nTiles = (M >> 4) * tilesN;
    if (tile >= nTiles) return;
    int tM = tile / tilesN;
    int tN = tile % tilesN;
    int row = lane & 15;
    int hi  = lane >> 4;

    const _Float16* arow = A + (size_t)(tM * 16 + row) * Kp;
    const _Float16* brow = W + (size_t)(tN * 16 + row) * Kp;

    v8f acc = {};
    for (int kk = 0; kk < Kp; kk += 32) {
        const _Float16* ap = arow + kk + hi * 8;
        const _Float16* bp = brow + kk + hi * 8;
        v16h a, b;
#pragma unroll
        for (int t = 0; t < 8; ++t) { a[t] = ap[t]; a[t + 8] = ap[t + 16]; }
#pragma unroll
        for (int t = 0; t < 8; ++t) { b[t] = bp[t]; b[t + 8] = bp[t + 16]; }
        acc = __builtin_amdgcn_wmma_f32_16x16x32_f16(false, a, false, b,
                                                     (short)0, acc, false, false);
    }
#pragma unroll
    for (int t = 0; t < 8; ++t) {
        int mm = tM * 16 + hi * 8 + t;
        int nn = tN * 16 + row;
        float v = fmaxf(acc[t] + bias[nn], 0.f);
        if (STORE_F16) ((_Float16*)out)[(size_t)mm * N + nn] = (_Float16)v;
        else           ((float*)out)[(size_t)mm * N + nn] = v;
    }
}

// ---------------------------------------------------------------------------
// per-sample: L2-normalize 256 features, 8-qubit 7-layer RY/CNOT statevector,
// <Z0>, classifier MLP 1->128->64->32->16->1 with BN/ReLU + sigmoid.
// One 256-thread block per sample; state ping-pongs in LDS.
// ---------------------------------------------------------------------------
__global__ void quantum_classifier(const float* __restrict__ h2, const float* __restrict__ qw,
                                   const float* __restrict__ c1w, const float* __restrict__ c1b,
                                   const float* __restrict__ n1g, const float* __restrict__ n1b,
                                   const float* __restrict__ n1m, const float* __restrict__ n1v,
                                   const float* __restrict__ c2w, const float* __restrict__ c2b,
                                   const float* __restrict__ n2g, const float* __restrict__ n2b,
                                   const float* __restrict__ n2m, const float* __restrict__ n2v,
                                   const float* __restrict__ c3w, const float* __restrict__ c3b,
                                   const float* __restrict__ c4w, const float* __restrict__ c4b,
                                   const float* __restrict__ c5w, const float* __restrict__ c5b,
                                   float* __restrict__ out) {
    __shared__ float sA[256];
    __shared__ float sB[256];
    int tid = threadIdx.x;
    int b = blockIdx.x;

    float v = h2[(size_t)b * 256 + tid];
    sB[tid] = v * v;
    __syncthreads();
    for (int s = 128; s > 0; s >>= 1) {
        if (tid < s) sB[tid] += sB[tid + s];
        __syncthreads();
    }
    float inv = rsqrtf(fmaxf(sB[0], 1e-24f));
    __syncthreads();
    sA[tid] = v * inv;
    __syncthreads();

    float* cur = sA;
    float* nxt = sB;
    for (int layer = 0; layer < 7; ++layer) {
        for (int q = 0; q < 8; ++q) {
            float th = qw[layer * 8 + q] * 0.5f;
            float c = cosf(th), s = sinf(th);
            int mask = 1 << (7 - q);
            float self = cur[tid];
            float other = cur[tid ^ mask];
            nxt[tid] = (tid & mask) ? (c * self + s * other) : (c * self - s * other);
            __syncthreads();
            float* tmp = cur; cur = nxt; nxt = tmp;
        }
        for (int q = 0; q < 7; ++q) {
            int bc = 7 - q, bt = 6 - q;
            int src = ((tid >> bc) & 1) ? (tid ^ (1 << bt)) : tid;
            nxt[tid] = cur[src];
            __syncthreads();
            float* tmp = cur; cur = nxt; nxt = tmp;
        }
    }

    float p = cur[tid] * cur[tid];
    nxt[tid] = (tid < 128) ? p : -p;
    __syncthreads();
    for (int s = 128; s > 0; s >>= 1) {
        if (tid < s) nxt[tid] += nxt[tid + s];
        __syncthreads();
    }
    float z = nxt[0];
    __syncthreads();

    if (tid < 128) {
        float t = fmaxf(z * c1w[tid] + c1b[tid], 0.f);
        float iv = n1g[tid] * rsqrtf(n1v[tid] + BN_EPS);
        sA[tid] = t * iv + (n1b[tid] - n1m[tid] * iv);
    }
    __syncthreads();
    if (tid < 64) {
        float acc = c2b[tid];
        for (int k = 0; k < 128; ++k) acc += sA[k] * c2w[tid * 128 + k];
        float t = fmaxf(acc, 0.f);
        float iv = n2g[tid] * rsqrtf(n2v[tid] + BN_EPS);
        sB[tid] = t * iv + (n2b[tid] - n2m[tid] * iv);
    }
    __syncthreads();
    if (tid < 32) {
        float acc = c3b[tid];
        for (int k = 0; k < 64; ++k) acc += sB[k] * c3w[tid * 64 + k];
        sA[tid] = fmaxf(acc, 0.f);
    }
    __syncthreads();
    if (tid < 16) {
        float acc = c4b[tid];
        for (int k = 0; k < 32; ++k) acc += sA[k] * c4w[tid * 32 + k];
        sB[tid] = fmaxf(acc, 0.f);
    }
    __syncthreads();
    if (tid == 0) {
        float acc = c5b[0];
        for (int k = 0; k < 16; ++k) acc += sB[k] * c5w[k];
        out[b] = 1.f / (1.f + expf(-acc));
    }
}

// ---------------------------------------------------------------------------
extern "C" void kernel_launch(void* const* d_in, const int* in_sizes, int n_in,
                              void* d_out, int out_size, void* d_ws, size_t ws_size,
                              hipStream_t stream) {
    const float* x       = (const float*)d_in[0];
    const float* conv1_w = (const float*)d_in[1];  const float* conv1_b = (const float*)d_in[2];
    const float* bn1_g = (const float*)d_in[3], *bn1_b = (const float*)d_in[4];
    const float* bn1_m = (const float*)d_in[5], *bn1_v = (const float*)d_in[6];
    const float* conv2_w = (const float*)d_in[7];  const float* conv2_b = (const float*)d_in[8];
    const float* bn2_g = (const float*)d_in[9], *bn2_b = (const float*)d_in[10];
    const float* bn2_m = (const float*)d_in[11], *bn2_v = (const float*)d_in[12];
    const float* conv3_w = (const float*)d_in[13]; const float* conv3_b = (const float*)d_in[14];
    const float* bn3_g = (const float*)d_in[15], *bn3_b = (const float*)d_in[16];
    const float* bn3_m = (const float*)d_in[17], *bn3_v = (const float*)d_in[18];
    const float* fc1_w = (const float*)d_in[19], *fc1_b = (const float*)d_in[20];
    const float* fc2_w = (const float*)d_in[21], *fc2_b = (const float*)d_in[22];
    const float* qw    = (const float*)d_in[23];
    const float* c1_w = (const float*)d_in[24], *c1_b = (const float*)d_in[25];
    const float* cbn1_g = (const float*)d_in[26], *cbn1_b = (const float*)d_in[27];
    const float* cbn1_m = (const float*)d_in[28], *cbn1_v = (const float*)d_in[29];
    const float* c2_w = (const float*)d_in[30], *c2_b = (const float*)d_in[31];
    const float* cbn2_g = (const float*)d_in[32], *cbn2_b = (const float*)d_in[33];
    const float* cbn2_m = (const float*)d_in[34], *cbn2_v = (const float*)d_in[35];
    const float* c3_w = (const float*)d_in[36], *c3_b = (const float*)d_in[37];
    const float* c4_w = (const float*)d_in[38], *c4_b = (const float*)d_in[39];
    const float* c5_w = (const float*)d_in[40], *c5_b = (const float*)d_in[41];

    char* ws = (char*)d_ws;
    // workspace layout (bytes)
    _Float16* a1 = (_Float16*)(ws + 0);          // conv1 out f16 [1024*1024][12] = 25,165,824 B
    _Float16* a2 = (_Float16*)(ws + 25165824);   // conv2 out f16 [262144][24]    = 12,582,912 B
    float*    a3 = (float*)(ws + 37748736);      // conv3 out f32 [262144][48]    = 50,331,648 B
    char*     sm = ws + 88080384;
    _Float16* featH = (_Float16*)(sm);                       // 1024*448 f16 = 917,504
    _Float16* w1h   = (_Float16*)(sm + 917504);              // 512*448  f16 = 458,752
    _Float16* h1h   = (_Float16*)(sm + 1376256);             // 1024*512 f16 = 1,048,576
    _Float16* w2h   = (_Float16*)(sm + 2424832);             // 256*512  f16 = 262,144
    _Float16* w2c   = (_Float16*)(sm + 2686976);             // 32*128   f16 = 8,192
    _Float16* w3c   = (_Float16*)(sm + 2695168);             // 48*224   f16 = 21,504
    float*    h2    = (float*)(sm + 2716672);                // 1024*256 f32 = 1,048,576

    // weight packing (independent of conv chain)
    cvt_f16_pad<<<896, 256, 0, stream>>>(fc1_w, w1h, 432, 448, 512 * 448);
    cvt_f16_pad<<<512, 256, 0, stream>>>(fc2_w, w2h, 512, 512, 256 * 512);
    pack_convw_f16<<<16, 256, 0, stream>>>(conv2_w, w2c, 108, 128, 24, 32 * 128);
    pack_convw_f16<<<42, 256, 0, stream>>>(conv3_w, w3c, 216, 224, 48, 48 * 224);

    // conv1 (VALU, reads the image) -> f16 activations
    conv1_bn_relu_f16<<<49152, 256, 0, stream>>>(x, conv1_w, conv1_b,
                                                 bn1_g, bn1_b, bn1_m, bn1_v, a1);
    // conv2 implicit-GEMM WMMA: M=262144, N=24(pad 32), K=108(pad 128); 32768 tiles
    conv_wmma_bn_relu<12, 32, 2, 16, 108, 128, 24, 32, true>
        <<<4096, 256, 0, stream>>>(a1, w2c, conv2_b, bn2_g, bn2_b, bn2_m, bn2_v, (void*)a2);
    // conv3 implicit-GEMM WMMA: M=262144, N=48, K=216(pad 224); 49152 tiles
    conv_wmma_bn_relu<24, 16, 1, 16, 216, 224, 48, 48, false>
        <<<6144, 256, 0, stream>>>(a2, w3c, conv3_b, bn3_g, bn3_b, bn3_m, bn3_v, (void*)a3);

    pool_flatten_f16<<<1792, 256, 0, stream>>>(a3, featH);

    // fc1: [1024,448]x[512,448]^T -> relu -> f16 ; 2048 tiles
    gemm_wmma_relu<true><<<256, 256, 0, stream>>>(featH, w1h, fc1_b, (void*)h1h, 1024, 512, 448);
    // fc2: [1024,512]x[256,512]^T -> relu -> f32 ; 1024 tiles
    gemm_wmma_relu<false><<<128, 256, 0, stream>>>(h1h, w2h, fc2_b, (void*)h2, 1024, 256, 512);

    // normalize + quantum circuit + classifier, one block per sample
    quantum_classifier<<<1024, 256, 0, stream>>>(h2, qw,
        c1_w, c1_b, cbn1_g, cbn1_b, cbn1_m, cbn1_v,
        c2_w, c2_b, cbn2_g, cbn2_b, cbn2_m, cbn2_v,
        c3_w, c3_b, c4_w, c4_b, c5_w, c5_b,
        (float*)d_out);
}
